// MultichannelMultiheadAttention2_7284264534706
// MI455X (gfx1250) — matrix operand order repaired and checked by reading, not compiled, not run on hardware
//
#include <hip/hip_runtime.h>
#include <hip/hip_bf16.h>
#include <cstdint>

typedef __bf16 bf16;
typedef __attribute__((ext_vector_type(16))) __bf16 v16bf;
typedef __attribute__((ext_vector_type(8)))  __bf16 v8bf;
typedef __attribute__((ext_vector_type(8)))  float  v8f;

// ---------------- problem dims ----------------
constexpr int NB = 2, NCH = 8, NAM = 8, NNH = 8, NF = 256, NW = 768, NDH = 32;
constexpr size_t OUT0 = (size_t)NB * NCH * NF * NW;            // 3,145,728 floats (out), qk follows

// ---------------- workspace layout (bytes) ----------------
constexpr size_t YT_ELE = (size_t)3 * NB * NAM * NW * NF;       // conv outputs, [qkv][b][am][w][f] bf16
constexpr size_t QKV_ELE = (size_t)NB * NAM * NNH * NW * NDH;   // per tensor
constexpr size_t XA_ELE = (size_t)NB * 16 * NW * NF;            // concat(x,a) transposed [b][c16][w][f] bf16
constexpr size_t O1_ELE = (size_t)NB * 16 * NF * NW;            // o_pw result fp32
constexpr size_t PW1_ELE = (size_t)NAM * NF * NF;               // 524288

constexpr size_t YT_OFF = 0;
constexpr size_t QT_OFF = YT_OFF + YT_ELE * 2;
constexpr size_t KT_OFF = QT_OFF + QKV_ELE * 2;
constexpr size_t VT_OFF = KT_OFF + QKV_ELE * 2;
constexpr size_t XA_OFF = VT_OFF + QKV_ELE * 2;
constexpr size_t O1_OFF = XA_OFF + XA_ELE * 2;
constexpr size_t PW_OFF = O1_OFF + O1_ELE * 4;                  // bf16 copies of q/k/v/o pw weights

// ---------------- WMMA fragment helpers (CDNA5 wave32 layouts) ----------------
// A 16x32 bf16: lanes 0-15 row M=lane hold K={0..7,16..23}; lanes 16-31 row M=lane-16 hold K={8..15,24..31}
__device__ __forceinline__ v16bf load_A16x32(const bf16* base, int ld, int lane) {
  int row = lane & 15, hi = lane >> 4;
  const bf16* p = base + row * ld + hi * 8;
  v8bf a = *(const v8bf*)(p);
  v8bf b = *(const v8bf*)(p + 16);
  v16bf r;
#pragma unroll
  for (int i = 0; i < 8; ++i) { r[i] = a[i]; r[i + 8] = b[i]; }
  return r;
}
// B 32x16 bf16 from memory stored [n][k] (row = column N, K contiguous):
// lanes 0-15 col N=lane hold K={0..7,16..23}; lanes 16-31 col N=lane-16 hold K={8..15,24..31}
__device__ __forceinline__ v16bf load_B32x16(const bf16* base, int ld, int lane) {
  int col = lane & 15, hi = lane >> 4;
  const bf16* p = base + col * ld + hi * 8;
  v8bf a = *(const v8bf*)(p);
  v8bf b = *(const v8bf*)(p + 16);
  v16bf r;
#pragma unroll
  for (int i = 0; i < 8; ++i) { r[i] = a[i]; r[i + 8] = b[i]; }
  return r;
}
__device__ __forceinline__ v8f wmma_bf16(v16bf A, v16bf B, v8f C) {
  return __builtin_amdgcn_wmma_f32_16x16x32_bf16(false, A, false, B, (short)0, C, false, false);
}

// ---------------- kernel 0: fp32 -> bf16 weight conversion ----------------
__global__ void wcvt_kernel(const float* qpw, const float* kpw, const float* vpw,
                            const float* opw, bf16* dst) {
  size_t i = (size_t)blockIdx.x * 256 + threadIdx.x;
  if (i < PW1_ELE)               dst[i] = (bf16)qpw[i];
  else if (i < 2 * PW1_ELE)      dst[i] = (bf16)kpw[i - PW1_ELE];
  else if (i < 3 * PW1_ELE)      dst[i] = (bf16)vpw[i - 2 * PW1_ELE];
  else if (i < 5 * PW1_ELE)      dst[i] = (bf16)opw[i - 3 * PW1_ELE];   // o_pw is 16*256*256
}

// ---------------- kernel 1: 3x3 conv (pad 1) + bias, write y_t[qkv][b][am][w][f] bf16 ----------------
__global__ void conv_kernel(const float* __restrict__ x,
                            const float* qcw, const float* qcb,
                            const float* kcw, const float* kcb,
                            const float* vcw, const float* vcb,
                            bf16* __restrict__ y_t) {
  size_t idx = (size_t)blockIdx.x * 256 + threadIdx.x;
  if (idx >= (size_t)3 * NB * NAM * NF * NW) return;
  int w   = idx % NW;
  int f   = (idx / NW) % NF;
  int am  = (idx / ((size_t)NW * NF)) % NAM;
  int b   = (idx / ((size_t)NW * NF * NAM)) % NB;
  int qkv = idx / ((size_t)NW * NF * NAM * NB);
  const float* cw = qkv == 0 ? qcw : (qkv == 1 ? kcw : vcw);
  const float* cb = qkv == 0 ? qcb : (qkv == 1 ? kcb : vcb);
  float acc = cb[am];
#pragma unroll
  for (int ci = 0; ci < NCH; ++ci) {
    const float* xp = x + ((size_t)(b * NCH + ci) * NF) * NW;
    const float* wp = cw + (am * NCH + ci) * 9;
#pragma unroll
    for (int kh = 0; kh < 3; ++kh) {
      int ff = f + kh - 1;
      if (ff < 0 || ff >= NF) continue;
#pragma unroll
      for (int kw = 0; kw < 3; ++kw) {
        int ww = w + kw - 1;
        if (ww < 0 || ww >= NW) continue;
        acc += xp[(size_t)ff * NW + ww] * wp[kh * 3 + kw];
      }
    }
  }
  y_t[((size_t)(qkv * NB * NAM + b * NAM + am) * NW + w) * NF + f] = (bf16)acc;
}

// ---------------- kernel 2: transpose x into concat buffer channels 0..7 ----------------
__global__ void xpose_kernel(const float* __restrict__ x, bf16* __restrict__ xa_t) {
  size_t idx = (size_t)blockIdx.x * 256 + threadIdx.x;
  if (idx >= (size_t)NB * NCH * NF * NW) return;
  int w = idx % NW;
  int f = (idx / NW) % NF;
  int c = (idx / ((size_t)NW * NF)) % NCH;
  int b = idx / ((size_t)NW * NF * NCH);
  xa_t[((size_t)(b * 16 + c) * NW + w) * NF + f] =
      (bf16)x[((size_t)(b * NCH + c) * NF + f) * NW + w];
}

// ---------------- kernel 3: positionwise projection GEMM + fused RoPE ----------------
// one wave per 16x16 output tile; C = pw(256x256) @ y(256 x 768)
__global__ void proj_rope_kernel(const bf16* __restrict__ y_t,
                                 const bf16* __restrict__ qpw, const bf16* __restrict__ kpw,
                                 const bf16* __restrict__ vpw,
                                 bf16* __restrict__ q_t, bf16* __restrict__ k_t,
                                 bf16* __restrict__ v_t) {
  int wid  = blockIdx.x * 8 + (threadIdx.x >> 5);
  int lane = threadIdx.x & 31;
  int wt  = wid % 48;
  int gt  = (wid / 48) % 16;
  int am  = (wid / 768) % NAM;
  int b   = (wid / 6144) % NB;
  int qkv = wid / 12288;
  const bf16* pw = (qkv == 0) ? qpw : (qkv == 1 ? kpw : vpw);

  const bf16* Abase = pw + (size_t)am * NF * NF + (size_t)gt * 16 * NF;
  const bf16* Bbase = y_t + ((size_t)(qkv * NB * NAM + b * NAM + am) * NW + wt * 16) * NF;

  v8f c = {};
#pragma unroll
  for (int f0 = 0; f0 < NF; f0 += 32) {
    if (f0 + 32 < NF) __builtin_prefetch(Bbase + f0 + 32, 0, 0);
    v16bf A = load_A16x32(Abase + f0, NF, lane);
    v16bf Bm = load_B32x16(Bbase + f0, NF, lane);
    c = wmma_bf16(A, Bm, c);
  }

  int hi = lane >> 4;
  int gbase = gt * 16 + hi * 8;              // row g of c[j] is gbase + j
  int w = wt * 16 + (lane & 15);             // column (sequence position)

  if (qkv < 2) {                             // RoPE on q, k: pairs (even g, odd g) are (c[2j], c[2j+1])
    float wseq = (float)w;
#pragma unroll
    for (int j = 0; j < 8; j += 2) {
      int g = gbase + j;
      int p = (g & (NDH - 1)) >> 1;          // pair index within head dim
      float inv = __expf(-(float)p * 0.575646273248511f);  // 10000^(-p/16)
      float s, ct;
      __sincosf(wseq * inv, &s, &ct);
      float e = c[j], o = c[j + 1];
      c[j]     = e * ct - o * s;
      c[j + 1] = o * ct + e * s;
    }
  }

#pragma unroll
  for (int j = 0; j < 8; ++j) {
    int g = gbase + j;
    int h = g >> 5, d = g & (NDH - 1);
    size_t bamh = (size_t)(b * NAM + am) * NNH + h;
    if (qkv == 0)      q_t[(bamh * NW + w) * NDH + d] = (bf16)c[j];   // [w][d]
    else if (qkv == 1) k_t[(bamh * NW + w) * NDH + d] = (bf16)c[j];   // [w][d]
    else               v_t[(bamh * NDH + d) * NW + w] = (bf16)c[j];   // [d][w] for B-operand of P@V
  }
}

// ---------------- kernel 4: fused QK^T + prev_qk + qk-output + softmax + P@V ----------------
// block = 256 threads (8 waves) handles one (b, am, h, 16-row q block)
constexpr size_t SM_QK   = (size_t)16 * NW * 4;     // fp32 scores
constexpr size_t SM_P    = (size_t)16 * NW * 2;     // bf16 probabilities
constexpr size_t SM_RED  = 16 * 16 * 4;
constexpr size_t SM_ROW  = 16 * 4;
constexpr size_t SM_PART = (size_t)8 * 32 * 8 * 4;
constexpr size_t SMEM_BYTES = SM_QK + SM_P + SM_RED + 2 * SM_ROW + SM_PART;  // 83,072 B

__global__ void attn_kernel(const bf16* __restrict__ q_t, const bf16* __restrict__ k_t,
                            const bf16* __restrict__ v_t, const float* __restrict__ prev_qk,
                            float* __restrict__ qk_out, bf16* __restrict__ xa_t) {
  extern __shared__ __align__(16) char smem_raw[];
  float* qk_s   = (float*)(smem_raw);
  bf16*  p_s    = (bf16*)(smem_raw + SM_QK);
  float* red    = (float*)(smem_raw + SM_QK + SM_P);
  float* rowmax = (float*)(smem_raw + SM_QK + SM_P + SM_RED);
  float* rinv   = rowmax + 16;
  float* part   = rinv + 16;

  int bid = blockIdx.x;
  int qt = bid % 48;
  int h  = (bid / 48) % NNH;
  int am = (bid / 384) % NAM;
  int b  = bid / 3072;
  int q0 = qt * 16;

  int lane = threadIdx.x & 31;
  int wave = threadIdx.x >> 5;
  int hi = lane >> 4;

  size_t bamh = (size_t)(b * NAM + am) * NNH + h;
  const bf16* qbase = q_t + (bamh * NW + q0) * NDH;
  const bf16* kbase = k_t + bamh * NW * NDH;
  const bf16* vbase = v_t + bamh * NDH * NW;
  const float* prev = prev_qk + (bamh * NW + q0) * NW;
  float* qko        = qk_out + (bamh * NW + q0) * NW;

  // ---- scores: one K=32 WMMA per 16x16 tile; each wave owns 6 k-tiles ----
  v16bf Aq = load_A16x32(qbase, NDH, lane);
#pragma unroll
  for (int i = 0; i < 6; ++i) {
    int kt = wave * 6 + i;
    if (i + 1 < 6) __builtin_prefetch(kbase + (size_t)(kt + 1) * 16 * NDH, 0, 0);
    v16bf Bk = load_B32x16(kbase + (size_t)kt * 16 * NDH, NDH, lane);
    v8f zero = {};
    v8f c = wmma_bf16(Aq, Bk, zero);
    int n = kt * 16 + (lane & 15);
#pragma unroll
    for (int j = 0; j < 8; ++j) {
      int m = j + hi * 8;
      float val = c[j] * 0.0625f + prev[(size_t)m * NW + n];   // 1/sqrt(256)
      qko[(size_t)m * NW + n] = val;                            // returned qk tensor
      qk_s[m * NW + n] = val;
    }
  }
  __syncthreads();

  // ---- softmax over k (768) for 16 rows ----
  {
    int r = threadIdx.x >> 4, c0 = threadIdx.x & 15;
    float m = -1e30f;
    for (int i = c0; i < NW; i += 16) m = fmaxf(m, qk_s[r * NW + i]);
    red[r * 16 + c0] = m;
    __syncthreads();
    if (threadIdx.x < 16) {
      float mm = red[threadIdx.x * 16];
      for (int i = 1; i < 16; ++i) mm = fmaxf(mm, red[threadIdx.x * 16 + i]);
      rowmax[threadIdx.x] = mm;
    }
    __syncthreads();
    float mm = rowmax[r], s = 0.f;
    for (int i = c0; i < NW; i += 16) {
      float e = __expf(qk_s[r * NW + i] - mm);
      p_s[r * NW + i] = (bf16)e;
      s += e;
    }
    red[r * 16 + c0] = s;
    __syncthreads();
    if (threadIdx.x < 16) {
      float ss = 0.f;
      for (int i = 0; i < 16; ++i) ss += red[threadIdx.x * 16 + i];
      rinv[threadIdx.x] = 1.0f / ss;
    }
    __syncthreads();
  }

  // ---- P @ V : 24 K-chunks x 2 d-halves; wave -> (half = w&1, quarter = w>>1) ----
  {
    int half = wave & 1, quarter = wave >> 1;
    const bf16* vb = vbase + (size_t)half * 16 * NW;
    v8f acc = {};
#pragma unroll
    for (int i = 0; i < 6; ++i) {
      int chunk = quarter * 6 + i;
      v16bf Ap = load_A16x32(p_s + chunk * 32, NW, lane);
      v16bf Bv = load_B32x16(vb + chunk * 32, NW, lane);
      acc = wmma_bf16(Ap, Bv, acc);
    }
#pragma unroll
    for (int j = 0; j < 8; ++j) part[((size_t)wave * 32 + lane) * 8 + j] = acc[j];
    __syncthreads();
    if (wave < 2) {                       // wave 0 reduces half 0, wave 1 reduces half 1
      int h2 = wave;
#pragma unroll
      for (int j = 0; j < 8; ++j) {
        float s = part[((size_t)(h2 + 0) * 32 + lane) * 8 + j]
                + part[((size_t)(h2 + 2) * 32 + lane) * 8 + j]
                + part[((size_t)(h2 + 4) * 32 + lane) * 8 + j]
                + part[((size_t)(h2 + 6) * 32 + lane) * 8 + j];
        int m = j + hi * 8;
        float val = s * rinv[m];
        int f = h * NDH + h2 * 16 + (lane & 15);
        int w = q0 + m;
        xa_t[((size_t)(b * 16 + NCH + am) * NW + w) * NF + f] = (bf16)val;
      }
    }
  }
}

// ---------------- kernel 5: output projection GEMM (o_pw) ----------------
__global__ void opw_kernel(const bf16* __restrict__ xa_t, const bf16* __restrict__ opw,
                           float* __restrict__ out1) {
  int wid  = blockIdx.x * 8 + (threadIdx.x >> 5);
  int lane = threadIdx.x & 31;
  int wt = wid % 48;
  int gt = (wid / 48) % 16;
  int c  = (wid / 768) % 16;
  int b  = wid / 12288;

  const bf16* Abase = opw + (size_t)c * NF * NF + (size_t)gt * 16 * NF;
  const bf16* Bbase = xa_t + ((size_t)(b * 16 + c) * NW + wt * 16) * NF;

  v8f acc = {};
#pragma unroll
  for (int f0 = 0; f0 < NF; f0 += 32) {
    if (f0 + 32 < NF) __builtin_prefetch(Bbase + f0 + 32, 0, 0);
    v16bf A = load_A16x32(Abase + f0, NF, lane);
    v16bf Bm = load_B32x16(Bbase + f0, NF, lane);
    acc = wmma_bf16(A, Bm, acc);
  }
  int hi = lane >> 4;
  int n = wt * 16 + (lane & 15);
  float* obase = out1 + ((size_t)(b * 16 + c) * NF + gt * 16) * NW;
#pragma unroll
  for (int j = 0; j < 8; ++j) {
    int m = j + hi * 8;
    obase[(size_t)m * NW + n] = acc[j];
  }
}

// ---------------- kernel 6: depthwise channel mix (o_dw) ----------------
__global__ void odw_kernel(const float* __restrict__ out1, const float* __restrict__ o_dw,
                           float* __restrict__ out) {
  size_t idx = (size_t)blockIdx.x * 256 + threadIdx.x;
  if (idx >= (size_t)NB * NF * NW) return;
  int w = idx % NW;
  int g = (idx / NW) % NF;
  int b = idx / ((size_t)NW * NF);
  float sums[NCH];
#pragma unroll
  for (int d = 0; d < NCH; ++d) sums[d] = 0.f;
  for (int c = 0; c < 16; ++c) {
    float v = out1[((size_t)(b * 16 + c) * NF + g) * NW + w];
#pragma unroll
    for (int d = 0; d < NCH; ++d) sums[d] += o_dw[d * 16 + c] * v;
  }
#pragma unroll
  for (int d = 0; d < NCH; ++d)
    out[((size_t)(b * NCH + d) * NF + g) * NW + w] = sums[d];
}

// ---------------- host launch ----------------
extern "C" void kernel_launch(void* const* d_in, const int* in_sizes, int n_in,
                              void* d_out, int out_size, void* d_ws, size_t ws_size,
                              hipStream_t stream) {
  const float* x    = (const float*)d_in[0];
  const float* prev = (const float*)d_in[1];
  const float* q_cw = (const float*)d_in[2];
  const float* q_cb = (const float*)d_in[3];
  const float* q_pw = (const float*)d_in[4];
  const float* k_cw = (const float*)d_in[5];
  const float* k_cb = (const float*)d_in[6];
  const float* k_pw = (const float*)d_in[7];
  const float* v_cw = (const float*)d_in[8];
  const float* v_cb = (const float*)d_in[9];
  const float* v_pw = (const float*)d_in[10];
  const float* o_pw = (const float*)d_in[11];
  const float* o_dw = (const float*)d_in[12];

  char* ws = (char*)d_ws;
  bf16* y_t  = (bf16*)(ws + YT_OFF);
  bf16* q_t  = (bf16*)(ws + QT_OFF);
  bf16* k_t  = (bf16*)(ws + KT_OFF);
  bf16* v_t  = (bf16*)(ws + VT_OFF);
  bf16* xa_t = (bf16*)(ws + XA_OFF);
  float* out1 = (float*)(ws + O1_OFF);
  bf16* pwbf = (bf16*)(ws + PW_OFF);
  bf16* qpw_bf = pwbf;
  bf16* kpw_bf = pwbf + PW1_ELE;
  bf16* vpw_bf = pwbf + 2 * PW1_ELE;
  bf16* opw_bf = pwbf + 3 * PW1_ELE;

  float* out    = (float*)d_out;
  float* qk_out = out + OUT0;

  // 0) weight fp32->bf16 (5*524288 elements)
  wcvt_kernel<<<(5 * PW1_ELE + 255) / 256, 256, 0, stream>>>(q_pw, k_pw, v_pw, o_pw, pwbf);
  // 1) conv (3*2*8*256*768 = 9,437,184 threads)
  conv_kernel<<<36864, 256, 0, stream>>>(x, q_cw, q_cb, k_cw, k_cb, v_cw, v_cb, y_t);
  // 2) x transpose into concat buffer
  xpose_kernel<<<12288, 256, 0, stream>>>(x, xa_t);
  // 3) pw projection + RoPE: 3*2*8*16*48 = 36864 wave-tiles, 8 waves/block
  proj_rope_kernel<<<4608, 256, 0, stream>>>(y_t, qpw_bf, kpw_bf, vpw_bf, q_t, k_t, v_t);
  // 4) fused attention: 2*8*8*48 = 6144 blocks, 83 KB dynamic LDS
  attn_kernel<<<6144, 256, SMEM_BYTES, stream>>>(q_t, k_t, v_t, prev, qk_out, xa_t);
  // 5) output positionwise GEMM: 2*16*16*48 = 24576 wave-tiles
  opw_kernel<<<3072, 256, 0, stream>>>(xa_t, opw_bf, out1);
  // 6) depthwise channel mix
  odw_kernel<<<1536, 256, 0, stream>>>(out1, o_dw, out);
}
